// FactorizedSpectralConv_54254026883237
// MI455X (gfx1250) — matrix operand order repaired and checked by reading, not compiled, not run on hardware
//
#include <hip/hip_runtime.h>
#include <hip/hip_bf16.h>
#include <math.h>
#include <stdint.h>

typedef __attribute__((ext_vector_type(16))) _Float16 v16h;
typedef __attribute__((ext_vector_type(8)))  float    v8f;
typedef __attribute__((ext_vector_type(4)))  unsigned int u32x4;
typedef __attribute__((ext_vector_type(8)))  unsigned int u32x8;

#define BATCH 8
#define S 256          // spatial size
#define C 64           // channels
#define RK 32          // mode slots per dim (dim1 uses 17, rest zero-padded)
#define HID 128        // MLP hidden width

// ---------------- WMMA fragment helpers (wave32, 16x16x32 f16) ----------------
// A-matrix 16x32 f16 layout per CDNA5 ISA: lane L(0..15) holds row L, K in
// {0..7}U{16..23}; lane L+16 holds row L, K in {8..15}U{24..31}.
// Elements e=0..7 and e=8..15 are contiguous in LDS -> lowers to 2x ds_load_b128.
__device__ __forceinline__ v16h frag_a(const _Float16* base, int ld, int lane) {
  int h = lane >> 4, r = lane & 15;
  v16h a;
#pragma unroll
  for (int e = 0; e < 16; ++e) {
    int k = (e < 8 ? e : e + 8) + 8 * h;
    a[e] = base[r * ld + k];
  }
  return a;
}
// B fragments are staged pre-swizzled: frag f, lane l at sw[f*32 + l] (32B aligned),
// so a B fragment is one aligned v16h LDS load (2x ds_load_b128, no packing movs).
__device__ __forceinline__ int bsw_dst(int k, int n, int ntiles) {
  int kk = k >> 5, h = (k >> 3) & 1, e = (k & 7) + (((k >> 4) & 1) << 3);
  int tn = n >> 4, nl = n & 15;
  return (((kk * ntiles + tn) << 5) + (h << 4) + nl) * 16 + e;
}
__device__ __forceinline__ v8f wmma_f16(v16h a, v16h b, v8f c) {
  return __builtin_amdgcn_wmma_f32_16x16x32_f16(false, a, false, b, (short)0, c,
                                                false, false);
}

__device__ __forceinline__ float gelu_tanh(float x) {
  float x3 = x * x * x;
  return 0.5f * x * (1.0f + tanhf(0.7978845608028654f * (x + 0.044715f * x3)));
}

// ---------------- Tensor Data Mover: 2D f16 tile -> LDS (CDNA5 TDM) ----------------
// Builds D# groups 0/1 per CDNA5 ISA 8.3/8.4 (2-group form, <=2D tensor, VADDR2/3=NULL)
// and issues tensor_load_to_lds. Optional per-row LDS padding reproduces the padded
// row strides used by frag_a. Issue from one wave, then s_wait_tensorcnt + barrier.
__device__ __forceinline__ void tdm_load_2d_f16(const _Float16* gsrc, void* ldst,
                                                unsigned rows, unsigned cols,
                                                unsigned gstride,
                                                unsigned pad_interval_code,
                                                unsigned pad_amount_code,
                                                unsigned pad_en) {
  unsigned long long ga = (unsigned long long)(uintptr_t)gsrc;
  unsigned lds_addr = (unsigned)(uintptr_t)ldst;  // low 32 bits = LDS byte offset
  u32x4 g0 = {
      1u,                                              // count=1, user descriptor
      lds_addr,                                        // lds_addr
      (unsigned)(ga & 0xffffffffu),                    // global_addr[31:0]
      (unsigned)((ga >> 32) & 0x01ffffffu) | (2u << 30) // global_addr[56:32] | type=2
  };
  u32x8 g1 = {
      (1u << 16) | (pad_en << 20) | (pad_interval_code << 22) | (pad_amount_code << 25),
      (cols & 0xffffu) << 16,                          // tensor_dim0[15:0] (abar=0)
      ((cols >> 16) & 0xffffu) | ((rows & 0xffffu) << 16), // tensor_dim0 hi | tensor_dim1 lo
      ((rows >> 16) & 0xffffu) | ((cols & 0xffffu) << 16), // tensor_dim1 hi | tile_dim0
      (rows & 0xffffu),                                // tile_dim1 | tile_dim2=0
      gstride,                                         // tensor_dim0_stride[31:0]
      0u, 0u
  };
  asm volatile("tensor_load_to_lds %0, %1" :: "s"(g0), "s"(g1) : "memory");
}
__device__ __forceinline__ void tdm_wait() {
#if __has_builtin(__builtin_amdgcn_s_wait_tensorcnt)
  __builtin_amdgcn_s_wait_tensorcnt(0);
#else
  asm volatile("s_wait_tensorcnt 0x0" ::: "memory");
#endif
}

// ---------------- prep kernels ----------------
__global__ void k_build_dft(_Float16* Df0, _Float16* Df1, _Float16* Di0, _Float16* Di1) {
  int idx = blockIdx.x * blockDim.x + threadIdx.x;  // 4 * 16384
  int table = idx >> 14;
  int j = idx & 16383;
  const float TWO_PI_N = 6.283185307179586f / 256.0f;
  if (table < 2) {
    int m = j >> 8, x = j & 255;
    int r = m & 31; bool im = m >= 32;
    int dim = table;
    float v = 0.f;
    bool valid = (dim == 0) ? true : (r < 17);
    if (valid) {
      int f = (dim == 0) ? (r < 16 ? r : r + 97) : r;   // dim0 keeps freqs 0..15,113..128
      float ang = TWO_PI_N * (float)((f * x) & 255);
      v = (im ? -sinf(ang) : cosf(ang)) * (1.0f / 256.0f);
    }
    (table == 0 ? Df0 : Df1)[j] = (_Float16)v;
  } else {
    int x = j >> 6, m = j & 63;
    int r = m & 31; bool im = m >= 32;
    int dim = table - 2;
    float v = 0.f;
    bool valid = (dim == 0) ? true : (r < 17);
    if (valid) {
      int f = (dim == 0) ? (r < 16 ? r : r + 97) : r;
      float w = (f == 0 || f == 128) ? 1.0f : 2.0f;     // Hermitian doubling
      float ang = TWO_PI_N * (float)((f * x) & 255);
      v = w * (im ? -sinf(ang) : cosf(ang));
    }
    (dim == 0 ? Di0 : Di1)[j] = (_Float16)v;
  }
}

// Per-mode complex mix as real block matrix [[Kre,-Kim],[Kim,Kre]] -> Kw[r][128][128] f16
__global__ void k_build_kw(const float* k0re, const float* k0im,
                           const float* k1re, const float* k1im,
                           _Float16* Kw0, _Float16* Kw1) {
  int total = 2 * RK * 128 * 128;
  for (int idx = blockIdx.x * blockDim.x + threadIdx.x; idx < total;
       idx += gridDim.x * blockDim.x) {
    int dim = idx / (RK * 128 * 128);
    int rem = idx % (RK * 128 * 128);
    int r = rem / (128 * 128);
    int mk = rem % (128 * 128);
    int m = mk >> 7, k = mk & 127;
    int i = m & 63, jc = k & 63;
    const float* re  = dim ? k1re : k0re;
    const float* imw = dim ? k1im : k0im;
    int nvalid = dim ? 17 : 32;
    float v = 0.f;
    if (r < nvalid) {
      float kre = re[(r * C + i) * C + jc];
      float kim = imw[(r * C + i) * C + jc];
      if (m < 64) v = (k < 64) ? kre : -kim;
      else        v = (k < 64) ? kim :  kre;
    }
    (dim ? Kw1 : Kw0)[rem] = (_Float16)v;
  }
}

__global__ void k_cvt_mlp(const float* W0, const float* W1, _Float16* W0h, _Float16* W1h) {
  int idx = blockIdx.x * blockDim.x + threadIdx.x;  // 16384
  if (idx < C * HID) W0h[idx] = (_Float16)W0[idx];
  else               W1h[idx - C * HID] = (_Float16)W1[idx - C * HID];
}

// ---------------- pass kernels ----------------
// Forward: Xf[p][64 rc][64 ch] = Dfwd[64][256] @ Xcol[256][64], p=(b,q), 2048 blocks
__global__ void __launch_bounds__(256) k_fwd(const float* X, const _Float16* Dfwd,
                                             _Float16* Xf, int dim) {
  __shared__ _Float16 sD[64 * 136];
  __shared__ v16h sXsw[512];                 // 16 frags (kk0..3 x tn0..3) x 32 lanes
  _Float16* sx = (_Float16*)sXsw;
  int tid = threadIdx.x, lane = tid & 31, wave = tid >> 5;
  int p = blockIdx.x;
  int b = p >> 8, q = p & 255;
  long base = (long)b * (S * S * C) + (dim == 0 ? (long)q * C : (long)q * (S * C));
  long strideT = (dim == 0) ? (long)(S * C) : (long)C;

  v8f acc0 = {}; v8f acc1 = {};
  int t0 = wave * 2, t1 = t0 + 1;
  int tm0 = t0 >> 2, tn0 = t0 & 3, tm1 = t1 >> 2, tn1 = t1 & 3;

  for (int kc = 0; kc < 2; ++kc) {           // K = 256 in two 128 chunks
    // D chunk [64][128] via TDM: rows=64, cols=128 (64 DW) + 4 DW pad -> ld 136
    if (wave == 0)
      tdm_load_2d_f16(Dfwd + kc * 128, sD, 64, 128, 256, /*pi 64DW*/5, /*pa 4DW*/3, 1);
    // source-major (coalesced global reads), scatter into swizzled LDS
    for (int i = tid; i < 128 * 64; i += 256) {
      int t = i >> 6, n = i & 63;
      sx[bsw_dst(t, n, 4)] = (_Float16)X[base + (long)(kc * 128 + t) * strideT + n];
    }
    tdm_wait();
    __syncthreads();
#pragma unroll
    for (int kk = 0; kk < 4; ++kk) {
      v16h a0 = frag_a(&sD[(tm0 * 16) * 136 + kk * 32], 136, lane);
      v16h b0v = sXsw[(kk * 4 + tn0) * 32 + lane];
      acc0 = wmma_f16(a0, b0v, acc0);
      v16h a1 = frag_a(&sD[(tm1 * 16) * 136 + kk * 32], 136, lane);
      v16h b1v = sXsw[(kk * 4 + tn1) * 32 + lane];
      acc1 = wmma_f16(a1, b1v, acc1);
    }
    __syncthreads();
  }
  int h = lane >> 4, col = lane & 15;
  _Float16* out = Xf + (long)p * 64 * 64;
#pragma unroll
  for (int e = 0; e < 8; ++e) {
    out[(tm0 * 16 + e + 8 * h) * 64 + tn0 * 16 + col] = (_Float16)acc0[e];
    out[(tm1 * 16 + e + 8 * h) * 64 + tn1 * 16 + col] = (_Float16)acc1[e];
  }
}

// Mix: per mode r, Y = Kw[r](128x128) @ B(128 x 128 positions); grid (32 modes, 16 pos-blocks)
__global__ void __launch_bounds__(256) k_mix(const _Float16* Kw, const _Float16* Xf,
                                             _Float16* Y) {
  __shared__ _Float16 sK[128 * 72];
  __shared__ v16h sBsw[512];                 // 16 frags (kk0..1 x tn0..7) x 32 lanes
  _Float16* sb = (_Float16*)sBsw;
  int tid = threadIdx.x, lane = tid & 31, wave = tid >> 5;
  int r = blockIdx.x;
  int p0 = blockIdx.y * 128;
  const _Float16* Kr = Kw + (long)r * 128 * 128;

  v8f zero = {};
  v8f acc[8];
#pragma unroll
  for (int j = 0; j < 8; ++j) acc[j] = zero;

  for (int kc = 0; kc < 2; ++kc) {           // K = 128 in two 64 chunks (re then im)
    // K chunk [128][64] via TDM: rows=128, cols=64 (32 DW) + 4 DW pad -> ld 72
    if (wave == 0)
      tdm_load_2d_f16(Kr + kc * 64, sK, 128, 64, 128, /*pi 32DW*/4, /*pa 4DW*/3, 1);
    int rc_in = (kc == 0) ? r : (32 + r);
    for (int i = tid; i < 128 * 64; i += 256) {
      int pl = i >> 6, kl = i & 63;
      sb[bsw_dst(kl, pl, 8)] = Xf[((long)(p0 + pl) * 64 + rc_in) * 64 + kl];
    }
    tdm_wait();
    __syncthreads();
#pragma unroll
    for (int j = 0; j < 8; ++j) {
      int t = wave * 8 + j, tm = t >> 3, tn = t & 7;
#pragma unroll
      for (int kk = 0; kk < 2; ++kk) {
        v16h a = frag_a(&sK[(tm * 16) * 72 + kk * 32], 72, lane);
        v16h bv = sBsw[(kk * 8 + tn) * 32 + lane];
        acc[j] = wmma_f16(a, bv, acc[j]);
      }
    }
    __syncthreads();
  }
  int h = lane >> 4, col = lane & 15;
#pragma unroll
  for (int j = 0; j < 8; ++j) {
    int t = wave * 8 + j, tm = t >> 3, tn = t & 7;
#pragma unroll
    for (int e = 0; e < 8; ++e) {
      int m = tm * 16 + e + 8 * h;
      int pl = tn * 16 + col;
      int rc = (m < 64) ? r : (32 + r);
      int c = m & 63;
      Y[((long)(p0 + pl) * 64 + rc) * 64 + c] = (_Float16)acc[j][e];
    }
  }
}

// Inverse: Acc[t=0..255][ch] (+)= Dinv[256][64] @ Y[p][64][64]; 2048 blocks
__global__ void __launch_bounds__(256) k_inv(const _Float16* Dinv, const _Float16* Y,
                                             _Float16* Acc, int dim, int accumulate) {
  __shared__ _Float16 sD[256 * 72];
  __shared__ v16h sYsw[256];                 // 8 frags (kk0..1 x tn0..3) x 32 lanes
  _Float16* sy = (_Float16*)sYsw;
  int tid = threadIdx.x, lane = tid & 31, wave = tid >> 5;
  int p = blockIdx.x;
  int b = p >> 8, q = p & 255;

  // Dinv [256][64] via TDM: rows=256, cols=64 (32 DW) + 4 DW pad -> ld 72
  if (wave == 0)
    tdm_load_2d_f16(Dinv, sD, 256, 64, 64, /*pi 32DW*/4, /*pa 4DW*/3, 1);
  for (int i = tid; i < 64 * 64; i += 256) {
    int m = i >> 6, c = i & 63;
    sy[bsw_dst(m, c, 4)] = Y[((long)p * 64 + m) * 64 + c];
  }
  tdm_wait();
  __syncthreads();

  long bbase = (long)b * (S * S * C);
  int h = lane >> 4, col = lane & 15;
#pragma unroll
  for (int j = 0; j < 8; ++j) {
    int tidx = wave * 8 + j, tm = tidx >> 2, tn = tidx & 3;
    v8f acc = {};
#pragma unroll
    for (int kk = 0; kk < 2; ++kk) {
      v16h a = frag_a(&sD[(tm * 16) * 72 + kk * 32], 72, lane);
      v16h bv = sYsw[(kk * 4 + tn) * 32 + lane];
      acc = wmma_f16(a, bv, acc);
    }
#pragma unroll
    for (int e = 0; e < 8; ++e) {
      int t = tm * 16 + e + 8 * h;
      int i = tn * 16 + col;
      long idx = (dim == 0) ? (bbase + (long)t * (S * C) + (long)q * C + i)
                            : (bbase + (long)q * (S * C) + (long)t * C + i);
      float v = acc[e];
      if (accumulate) v += (float)Acc[idx];
      Acc[idx] = (_Float16)v;
    }
  }
}

// MLP: per 64-position tile, gelu(Xin@W0+b0)@W1+b1 -> f32 out; 8192 blocks
__global__ void __launch_bounds__(256) k_mlp(const _Float16* Acc, const _Float16* W0h,
                                             const float* b0, const _Float16* W1h,
                                             const float* b1, float* Out) {
  __shared__ _Float16 sXin[64 * 72];         // A for GEMM1
  __shared__ v16h sWsw[512];                 // W0 swizzled, then reused for W1
  __shared__ _Float16 sH[64 * 132];          // A for GEMM2
  _Float16* sw = (_Float16*)sWsw;
  int tid = threadIdx.x, lane = tid & 31, wave = tid >> 5;
  int p0 = blockIdx.x * 64;

  if (blockIdx.x + 1 < gridDim.x)
    __builtin_prefetch(Acc + (long)(p0 + 64) * 64, 0, 1);   // global_prefetch_b8

  for (int i = tid; i < 64 * 64; i += 256) {
    int pl = i >> 6, c = i & 63;
    sXin[pl * 72 + c] = Acc[(long)(p0 + pl) * 64 + c];
  }
  for (int i = tid; i < 64 * 128; i += 256) {             // W0: K=64, N=128, 8 n-tiles
    int cin = i >> 7, n = i & 127;
    sw[bsw_dst(cin, n, 8)] = W0h[i];
  }
  __syncthreads();

  int h = lane >> 4, col = lane & 15;
  // GEMM1: M=64 N=128 K=64 -> 32 tiles, 4 per wave
#pragma unroll
  for (int j = 0; j < 4; ++j) {
    int t = wave * 4 + j, tm = t >> 3, tn = t & 7;
    v8f acc = {};
#pragma unroll
    for (int kk = 0; kk < 2; ++kk) {
      v16h a = frag_a(&sXin[(tm * 16) * 72 + kk * 32], 72, lane);
      v16h bv = sWsw[(kk * 8 + tn) * 32 + lane];
      acc = wmma_f16(a, bv, acc);
    }
#pragma unroll
    for (int e = 0; e < 8; ++e) {
      int row = tm * 16 + e + 8 * h;
      int n = tn * 16 + col;
      sH[row * 132 + n] = (_Float16)gelu_tanh(acc[e] + b0[n]);
    }
  }
  __syncthreads();
  for (int i = tid; i < 128 * 64; i += 256) {             // W1: K=128, N=64, 4 n-tiles
    int n = i >> 6, o = i & 63;
    sw[bsw_dst(n, o, 4)] = W1h[i];
  }
  __syncthreads();
  // GEMM2: M=64 N=64 K=128 -> 16 tiles, 2 per wave
#pragma unroll
  for (int j = 0; j < 2; ++j) {
    int t = wave * 2 + j, tm = t >> 2, tn = t & 3;
    v8f acc = {};
#pragma unroll
    for (int kk = 0; kk < 4; ++kk) {
      v16h a = frag_a(&sH[(tm * 16) * 132 + kk * 32], 132, lane);
      v16h bv = sWsw[(kk * 4 + tn) * 32 + lane];
      acc = wmma_f16(a, bv, acc);
    }
#pragma unroll
    for (int e = 0; e < 8; ++e) {
      int row = tm * 16 + e + 8 * h;
      int o = tn * 16 + col;
      Out[(long)(p0 + row) * 64 + o] = acc[e] + b1[o];
    }
  }
}

extern "C" void kernel_launch(void* const* d_in, const int* in_sizes, int n_in,
                              void* d_out, int out_size, void* d_ws, size_t ws_size,
                              hipStream_t stream) {
  const float* X    = (const float*)d_in[0];
  const float* k0re = (const float*)d_in[1];
  const float* k0im = (const float*)d_in[2];
  const float* k1re = (const float*)d_in[3];
  const float* k1im = (const float*)d_in[4];
  const float* W0   = (const float*)d_in[5];
  const float* b0   = (const float*)d_in[6];
  const float* W1   = (const float*)d_in[7];
  const float* b1   = (const float*)d_in[8];
  float* Out = (float*)d_out;
  (void)in_sizes; (void)n_in; (void)out_size; (void)ws_size;

  char* ws = (char*)d_ws;
  size_t off = 0;
  auto take = [&](size_t bytes) {
    char* p = ws + off;
    off = (off + bytes + 255) & ~(size_t)255;
    return p;
  };
  _Float16* Df0 = (_Float16*)take(64 * 256 * 2);
  _Float16* Df1 = (_Float16*)take(64 * 256 * 2);
  _Float16* Di0 = (_Float16*)take(256 * 64 * 2);
  _Float16* Di1 = (_Float16*)take(256 * 64 * 2);
  _Float16* Kw0 = (_Float16*)take((size_t)RK * 128 * 128 * 2);
  _Float16* Kw1 = (_Float16*)take((size_t)RK * 128 * 128 * 2);
  _Float16* W0h = (_Float16*)take(C * HID * 2);
  _Float16* W1h = (_Float16*)take(HID * C * 2);
  _Float16* Xf  = (_Float16*)take((size_t)2048 * 64 * 64 * 2);
  _Float16* Yb  = (_Float16*)take((size_t)2048 * 64 * 64 * 2);
  _Float16* Acc = (_Float16*)take((size_t)BATCH * S * S * C * 2);

  k_build_dft<<<256, 256, 0, stream>>>(Df0, Df1, Di0, Di1);
  k_build_kw<<<2048, 256, 0, stream>>>(k0re, k0im, k1re, k1im, Kw0, Kw1);
  k_cvt_mlp<<<64, 256, 0, stream>>>(W0, W1, W0h, W1h);

  // dim 0 (along x)
  k_fwd<<<2048, 256, 0, stream>>>(X, Df0, Xf, 0);
  k_mix<<<dim3(32, 16), 256, 0, stream>>>(Kw0, Xf, Yb);
  k_inv<<<2048, 256, 0, stream>>>(Di0, Yb, Acc, 0, 0);

  // dim 1 (along y), reusing Xf/Yb scratch
  k_fwd<<<2048, 256, 0, stream>>>(X, Df1, Xf, 1);
  k_mix<<<dim3(32, 16), 256, 0, stream>>>(Kw1, Xf, Yb);
  k_inv<<<2048, 256, 0, stream>>>(Di1, Yb, Acc, 1, 1);

  k_mlp<<<8192, 256, 0, stream>>>(Acc, W0h, b0, W1h, b1, Out);
}
